// SSIM_2619930050927
// MI455X (gfx1250) — matrix operand order repaired and checked by reading, not compiled, not run on hardware
//
#include <hip/hip_runtime.h>

// CDNA5 WMMA vector types (probe-confirmed signatures)
typedef __attribute__((ext_vector_type(16))) __bf16 v16bf;
typedef __attribute__((ext_vector_type(8)))  __bf16 v8bf;
typedef __attribute__((ext_vector_type(8)))  float  v8f;

#define WSZ   11
#define IMH   512
#define IMW   512
#define NIMG  64

__device__ __forceinline__ v8f wmma_bf16(v16bf a, v16bf b, v8f c) {
  // D = A(16x32 bf16) x B(32x16 bf16) + C(16x16 f32)
  return __builtin_amdgcn_wmma_f32_16x16x32_bf16(
      /*neg_a=*/false, a, /*neg_b=*/false, b,
      /*c_mod=*/(short)0, c, /*reuse_a=*/false, /*reuse_b=*/false);
}

// Normalized 11-tap Gaussian (sigma = 1.5), tap j in [0,10]; 0 outside.
__device__ __forceinline__ float gauss_w(int j) {
  if (j < 0 || j >= WSZ) return 0.0f;
  float d = (float)(j - 5);
  // sum_{i=0..10} exp(-(i-5)^2/4.5) = 3.7592317
  return __expf(-d * d * (1.0f / 4.5f)) * (1.0f / 3.7592317f);
}

// f32 -> bf16 fragment builders (single rounding after f32 math)
__device__ __forceinline__ v16bf to_frag(const float* f) {
  v16bf r;
#pragma unroll
  for (int i = 0; i < 16; ++i) r[i] = (__bf16)f[i];
  return r;
}
__device__ __forceinline__ v16bf to_frag_mul(const float* a, const float* b) {
  v16bf r;
#pragma unroll
  for (int i = 0; i < 16; ++i) r[i] = (__bf16)(a[i] * b[i]);
  return r;
}

// Build the two vertical-pass B fragments (B[k][n] = In[y0-5+k][xbase+n]).
// CHECKED=true applies the conv zero-padding at image borders.
template <bool CHECKED>
__device__ __forceinline__ void load_inputs(const float* __restrict__ p1,
                                            const float* __restrict__ p2,
                                            int x0, int y0, int lm, int half,
                                            float f1[2][16], float f2[2][16]) {
#pragma unroll
  for (int hx = 0; hx < 2; ++hx) {
    const int col  = x0 - 5 + hx * 16 + lm;
    const int row0 = y0 - 5 + half * 16;
    if (CHECKED) {
      const bool cok = (col >= 0) && (col < IMW);
#pragma unroll
      for (int i = 0; i < 16; ++i) {
        const int row = row0 + i;
        const bool ok = cok && (row >= 0) && (row < IMH);
        const size_t idx = (size_t)row * IMW + col;
        f1[hx][i] = ok ? p1[idx] : 0.0f;
        f2[hx][i] = ok ? p2[idx] : 0.0f;
      }
    } else {
      const float* q1 = p1 + (size_t)row0 * IMW + col;
      const float* q2 = p2 + (size_t)row0 * IMW + col;
#pragma unroll
      for (int i = 0; i < 16; ++i) {
        f1[hx][i] = q1[(size_t)i * IMW];
        f2[hx][i] = q2[(size_t)i * IMW];
      }
    }
  }
}

__global__ void __launch_bounds__(128)
ssim_zero(float* out) {
  if (threadIdx.x == 0 && blockIdx.x == 0) out[0] = 0.0f;
}

__global__ void __launch_bounds__(128)
ssim_wmma_kernel(const float* __restrict__ img1,
                 const float* __restrict__ img2,
                 float* __restrict__ out) {
  // per-wave staging: 5 quantities x 16x32 bf16 vertical-blur tiles (row major)
  __shared__ __attribute__((aligned(16))) __bf16 tmp_lds[4][5 * 16 * 32];

  const int lane = threadIdx.x & 31;
  const int wave = threadIdx.x >> 5;
  const int half = lane >> 4;   // 0: lanes 0-15, 1: lanes 16-31
  const int lm   = lane & 15;

  // tile decomposition: 32x32 tiles per image, 64 images, 4 tiles per block
  const int tile          = blockIdx.x * 4 + wave;
  const int tiles_x       = IMW / 16;             // 32
  const int tiles_per_img = tiles_x * (IMH / 16); // 1024
  const int n   = tile / tiles_per_img;
  const int rem = tile % tiles_per_img;
  const int y0  = (rem / tiles_x) * 16;
  const int x0  = (rem % tiles_x) * 16;

  const size_t ioff = (size_t)n * IMH * IMW;
  const float* p1 = img1 + ioff;
  const float* p2 = img2 + ioff;

  // ---------------- constant weight fragments ----------------
  // Vertical: A = G (16x32), A[m][k] = w[k-m].
  // A-fragment layout: element i -> K = i + 16*half, M = lm
  v16bf gA;
#pragma unroll
  for (int i = 0; i < 16; ++i) {
    int K = i + 16 * half;
    gA[i] = (__bf16)gauss_w(K - lm);
  }
  // Horizontal: B = Gh (32x16), B[c][n] = w[c-n].
  // B-fragment layout: element i -> K = i + (i>=8?8:0) + 8*half, N = lm
  v16bf gB;
#pragma unroll
  for (int i = 0; i < 16; ++i) {
    int c = i + (i >= 8 ? 8 : 0) + 8 * half;
    gB[i] = (__bf16)gauss_w(c - lm);
  }

  // ---------------- input loads (f32 kept in registers) ----------------
  // Region columns x0-5 .. x0+26, rows y0-5 .. y0+26; zero padding at borders.
  float f1[2][16], f2[2][16];
  const bool interior = (x0 >= 16) && (x0 <= IMW - 32) &&
                        (y0 >= 16) && (y0 <= IMH - 32);
  if (interior) load_inputs<false>(p1, p2, x0, y0, lm, half, f1, f2);
  else          load_inputs<true >(p1, p2, x0, y0, lm, half, f1, f2);

  __bf16* myT = tmp_lds[wave];
  const v8f zacc = {};

  // ---------------- vertical blurs: 10 back-to-back WMMAs ----------------
  // quantities: 0:I1  1:I2  2:I1*I1  3:I2*I2  4:I1*I2  (products formed in f32)
#pragma unroll
  for (int q = 0; q < 5; ++q) {
    v16bf bb0, bb1;
    if      (q == 0) { bb0 = to_frag(f1[0]);            bb1 = to_frag(f1[1]);            }
    else if (q == 1) { bb0 = to_frag(f2[0]);            bb1 = to_frag(f2[1]);            }
    else if (q == 2) { bb0 = to_frag_mul(f1[0], f1[0]); bb1 = to_frag_mul(f1[1], f1[1]); }
    else if (q == 3) { bb0 = to_frag_mul(f2[0], f2[0]); bb1 = to_frag_mul(f2[1], f2[1]); }
    else             { bb0 = to_frag_mul(f1[0], f2[0]); bb1 = to_frag_mul(f1[1], f2[1]); }

    v8f t0 = wmma_bf16(gA, bb0, zacc);   // Tmp cols 0..15
    v8f t1 = wmma_bf16(gA, bb1, zacc);   // Tmp cols 16..31

    // D-layout -> row-major LDS: element v is (row v+8*half, col lm[+16])
    __bf16* T = myT + q * (16 * 32);
#pragma unroll
    for (int v = 0; v < 8; ++v) {
      T[(v + 8 * half) * 32 + lm]      = (__bf16)t0[v];
      T[(v + 8 * half) * 32 + lm + 16] = (__bf16)t1[v];
    }
  }

  __syncthreads();   // single barrier: all lane-crossing stores before reloads

  // ---------------- horizontal blurs: 5 WMMAs ----------------
  v8f blur[5];
#pragma unroll
  for (int q = 0; q < 5; ++q) {
    const __bf16* T = myT + q * (16 * 32);
    // A-fragment: (M = lm, K in {c0..c0+7} U {c0+16..c0+23}), c0 = 8*half.
    // Both chunks are contiguous, 16B-aligned -> two b128 LDS loads.
    const int c0 = 8 * half;
    v8bf lo = *(const v8bf*)(T + lm * 32 + c0);
    v8bf hi = *(const v8bf*)(T + lm * 32 + c0 + 16);
    v16bf aT = __builtin_shufflevector(lo, hi, 0, 1, 2, 3, 4, 5, 6, 7,
                                               8, 9, 10, 11, 12, 13, 14, 15);
    blur[q] = wmma_bf16(aT, gB, zacc);
  }

  // ---------------- SSIM map + mean ----------------
  const float C1c = 0.0001f;  // 0.01^2
  const float C2c = 0.0009f;  // 0.03^2
  float partial = 0.0f;
#pragma unroll
  for (int v = 0; v < 8; ++v) {
    float mu1 = blur[0][v];
    float mu2 = blur[1][v];
    float e11 = blur[2][v];
    float e22 = blur[3][v];
    float e12 = blur[4][v];
    float mu1s = mu1 * mu1;
    float mu2s = mu2 * mu2;
    float mu12 = mu1 * mu2;
    float s1  = e11 - mu1s;
    float s2  = e22 - mu2s;
    float s12 = e12 - mu12;
    float num = (2.0f * mu12 + C1c) * (2.0f * s12 + C2c);
    float den = (mu1s + mu2s + C1c) * (s1 + s2 + C2c);
    partial += num / den;
  }

  // wave32 reduction
#pragma unroll
  for (int off = 16; off > 0; off >>= 1)
    partial += __shfl_xor(partial, off, 32);

  if (lane == 0) {
    atomicAdd(out, partial * (1.0f / ((float)NIMG * IMH * IMW)));
  }
}

extern "C" void kernel_launch(void* const* d_in, const int* in_sizes, int n_in,
                              void* d_out, int out_size, void* d_ws, size_t ws_size,
                              hipStream_t stream) {
  (void)in_sizes; (void)n_in; (void)out_size; (void)d_ws; (void)ws_size;
  const float* img1 = (const float*)d_in[0];
  const float* img2 = (const float*)d_in[1];
  float* out = (float*)d_out;

  ssim_zero<<<1, 128, 0, stream>>>(out);

  const int tiles  = NIMG * (IMH / 16) * (IMW / 16); // 65536
  const int blocks = tiles / 4;                      // 16384 blocks x 4 waves
  ssim_wmma_kernel<<<blocks, 128, 0, stream>>>(img1, img2, out);
}
